// TATModel_70403103916665
// MI455X (gfx1250) — compile-verified, matrix operand order checked by reading, not run on hardware
//
#include <hip/hip_runtime.h>
#include <hip/hip_bf16.h>

typedef __attribute__((ext_vector_type(16))) _Float16 v16h;
typedef __attribute__((ext_vector_type(8)))  _Float16 v8h;
typedef __attribute__((ext_vector_type(4)))  _Float16 v4h;
typedef __attribute__((ext_vector_type(8)))  float    v8f;

#define LDA 136    // padded LDS leading dim (halves); 272B rows -> 16B-aligned 8-half chunks
#define LDAF 132   // padded LDS leading dim (floats) for f32 te tile; 528B rows, 16B-aligned
#define SLOPE 0.2f

// Swizzled W layout: 32 tiles (tk in 0..3, ct in 0..7; tile = tk*8+ct),
// per tile 32 lanes, per lane a BSLOT-half slot (first 16 halves valid = the
// v16h B fragment). 40B slot => ds_load_2addr_b64 pairs, lane stride 10 dwords
// => bank-conflict-free.
#define BSLOT 20
#define BTILE (32 * BSLOT)
#define WSW_HALVES (32 * BTILE)          // 20480 halves = 40KB
#define WSW_U4 (WSW_HALVES * 2 / 16)     // 2560 uint4

// ---------- fragment loaders ----------
__device__ inline v16h load_a_frag(const _Float16* __restrict__ A) {
  v8h a0 = *(const v8h*)(A);        // 16B aligned
  v8h a1 = *(const v8h*)(A + 16);   // 16B aligned
  return __builtin_shufflevector(a0, a1, 0,1,2,3,4,5,6,7,8,9,10,11,12,13,14,15);
}
__device__ inline v16h load_b_frag(const _Float16* __restrict__ sB, int tile, int lane) {
  const _Float16* base = sB + tile * BTILE + lane * BSLOT;
  v4h b0 = *(const v4h*)(base);
  v4h b1 = *(const v4h*)(base + 4);
  v4h b2 = *(const v4h*)(base + 8);
  v4h b3 = *(const v4h*)(base + 12);
  v8h lo = __builtin_shufflevector(b0, b1, 0,1,2,3,4,5,6,7);
  v8h hi = __builtin_shufflevector(b2, b3, 0,1,2,3,4,5,6,7);
  return __builtin_shufflevector(lo, hi, 0,1,2,3,4,5,6,7,8,9,10,11,12,13,14,15);
}

// ---------- order-preserving float<->uint for atomicMax ----------
__device__ inline unsigned enc_f(float f) {
  unsigned u = __float_as_uint(f);
  return (f >= 0.f) ? (u | 0x80000000u) : ~u;
}
__device__ inline float dec_f(unsigned u) {
  return (u & 0x80000000u) ? __uint_as_float(u & 0x7FFFFFFFu) : __uint_as_float(~u);
}

// ---------- prep: swizzled f16 weights + v_l + c0 ----------
__global__ void prep_kernel(const float* __restrict__ W,      // [256][128]
                            const float* __restrict__ att_l,  // [4][32]
                            const float* __restrict__ att_r,  // [4][32]
                            const float* __restrict__ phase,  // [128]
                            _Float16* __restrict__ WxSw,
                            _Float16* __restrict__ WtSw,
                            float* __restrict__ v_l,          // [4][128]
                            float* __restrict__ c0)           // [4]
{
  int t = threadIdx.x; // 128 threads
  for (int idx = t; idx < 32 * 32 * 16; idx += 128) {
    int tile = idx >> 9;
    int lane = (idx >> 4) & 31;
    int p = idx & 15;
    int tk = tile >> 3, ct = tile & 7;
    int hi = lane >> 4;
    int K = tk * 32 + (p & 7) + ((p >= 8) ? 16 : 0) + hi * 8;
    int n = ct * 16 + (lane & 15);
    int d = tile * BTILE + lane * BSLOT + p;
    WxSw[d] = (_Float16)W[K * 128 + n];
    WtSw[d] = (_Float16)W[(128 + K) * 128 + n];
  }
  for (int h = 0; h < 4; ++h) {
    float s = 0.f;
    for (int c = 0; c < 32; ++c) s += W[(128 + t) * 128 + h * 32 + c] * att_l[h * 32 + c];
    v_l[h * 128 + t] = s;
  }
  __shared__ float s_te0[128];
  __shared__ float s_t0w[128];
  s_te0[t] = cosf(phase[t]);
  __syncthreads();
  float acc = 0.f;
  for (int k = 0; k < 128; ++k) acc += s_te0[k] * W[(128 + k) * 128 + t];
  s_t0w[t] = acc;
  __syncthreads();
  if (t < 4) {
    float s = 0.f;
    for (int c = 0; c < 32; ++c) s += s_t0w[t * 32 + c] * att_r[t * 32 + c];
    c0[t] = s;
  }
}

// ---------- zero amax/esum/agg ----------
__global__ void zero3_kernel(unsigned* __restrict__ amax, float* __restrict__ esum,
                             float* __restrict__ agg, int nN) {
  int idx = blockIdx.x * blockDim.x + threadIdx.x;
  if (idx < nN * 4) { amax[idx] = 0u; esum[idx] = 0.f; }
  if (idx < nN * 128) agg[idx] = 0.f;
}

// ---------- xw = feat(f32) @ Wx(f16 swizzled), WMMA ----------
__global__ __launch_bounds__(256) void gemm_feat_kernel(
    const float* __restrict__ feat, const _Float16* __restrict__ WbSw,
    float* __restrict__ out, int nrows)
{
  __shared__ __align__(16) _Float16 sB[WSW_HALVES];  // 40KB
  __shared__ __align__(16) _Float16 sA[64 * LDA];    // 17.4KB
  int tid = threadIdx.x;
  int row0 = blockIdx.x * 64;
  const uint4* gB = (const uint4*)WbSw;
  uint4* lB = (uint4*)sB;
  for (int i = tid; i < WSW_U4; i += 256) lB[i] = gB[i];
  for (int idx = tid; idx < 64 * 128; idx += 256) {
    int r = idx >> 7, c = idx & 127;
    int gr = row0 + r;
    float v = (gr < nrows) ? feat[gr * 128 + c] : 0.f;
    sA[r * LDA + c] = (_Float16)v;
  }
  __syncthreads();
  int w = tid >> 5, lane = tid & 31;
  int rt = w & 3, ct0 = (w >> 2) * 4;
  int n = lane & 15, hi = (lane >> 4) & 1;
  const bool full = (row0 + 64 <= nrows);   // block-uniform tail guard
  const _Float16* aBase = sA + (rt * 16 + n) * LDA + hi * 8;
  v16h afr[4];
#pragma unroll
  for (int tk = 0; tk < 4; ++tk) afr[tk] = load_a_frag(aBase + tk * 32);
  for (int cti = 0; cti < 4; ++cti) {
    int ct = ct0 + cti;
    v8f acc = {};
#pragma unroll
    for (int tk = 0; tk < 4; ++tk) {
      v16h b = load_b_frag(sB, tk * 8 + ct, lane);
      acc = __builtin_amdgcn_wmma_f32_16x16x32_f16(false, afr[tk], false, b, (short)0, acc, false, false);
    }
    int col = ct * 16 + n;
    float* orow = out + (size_t)(row0 + rt * 16 + hi * 8) * 128 + col;
    if (full) {
#pragma unroll
      for (int q = 0; q < 8; ++q) orow[q * 128] = acc[q];
    } else {
#pragma unroll
      for (int q = 0; q < 8; ++q) {
        int row = row0 + rt * 16 + q + hi * 8;
        if (row < nrows) orow[q * 128] = acc[q];
      }
    }
  }
}

// ---------- per-node attention projections ----------
__global__ void attn_node_kernel(const float* __restrict__ xw,
                                 const float* __restrict__ att_l, const float* __restrict__ att_r,
                                 float* __restrict__ al, float* __restrict__ ar, int nN) {
  int idx = blockIdx.x * blockDim.x + threadIdx.x;
  if (idx >= nN * 4) return;
  int node = idx >> 2, h = idx & 3;
  const float* row = xw + node * 128 + h * 32;
  float sl = 0.f, sr = 0.f;
#pragma unroll
  for (int c = 0; c < 32; ++c) {
    float v = row[c];
    sl += v * att_l[h * 32 + c];
    sr += v * att_r[h * 32 + c];
  }
  al[idx] = sl; ar[idx] = sr;
}

// ---------- edge logits: a = lrelu(al[src] + te.v_l + ar[dst] + c0), atomicMax per dst ----------
__global__ __launch_bounds__(256) void edge_logits_kernel(
    const int* __restrict__ src, const int* __restrict__ dst, const float* __restrict__ ts,
    const float* __restrict__ freq, const float* __restrict__ phase,
    const float* __restrict__ v_l, const float* __restrict__ c0,
    const float* __restrict__ al, const float* __restrict__ ar,
    float* __restrict__ a_out, unsigned* __restrict__ amax, int nE)
{
  __shared__ __align__(16) float sTE[64 * LDAF];     // 33.8KB, f32 (no per-element cvt)
  __shared__ __align__(16) float sVL[4 * 128];
  __shared__ float sF[128], sP[128], sTs[64];
  int tid = threadIdx.x;
  int e0 = blockIdx.x * 64;
  if (tid < 128) { sF[tid] = freq[tid]; sP[tid] = phase[tid]; }
  for (int i = tid; i < 512; i += 256) sVL[i] = v_l[i];
  if (tid < 64) { int e = e0 + tid; sTs[tid] = (e < nE) ? ts[e] : 0.f; }
  __syncthreads();
  for (int idx = tid; idx < 64 * 128; idx += 256) {
    int le = idx >> 7, t = idx & 127;
    sTE[le * LDAF + t] = cosf(sTs[le] * sF[t] + sP[t]);
  }
  __syncthreads();
  int le = tid >> 2, h = tid & 3;
  int e = e0 + le;
  if (e < nE) {
    const float4* terow = (const float4*)(sTE + le * LDAF);
    const float4* vl = (const float4*)(sVL + h * 128);
    float s = 0.f;
#pragma unroll
    for (int t = 0; t < 32; ++t) {
      float4 a4 = terow[t], b4 = vl[t];
      s += a4.x * b4.x + a4.y * b4.y + a4.z * b4.z + a4.w * b4.w;
    }
    int sN = src[e], dN = dst[e];
    float a = al[sN * 4 + h] + s + ar[dN * 4 + h] + c0[h];
    a = (a > 0.f) ? a : a * SLOPE;
    a_out[e * 4 + h] = a;
    atomicMax(&amax[dN * 4 + h], enc_f(a));
  }
}

// ---------- softmax numerator + denominator ----------
__global__ void edge_exp_kernel(float* __restrict__ a_ea, const int* __restrict__ dst,
                                const unsigned* __restrict__ amax, float* __restrict__ esum, int nE) {
  int idx = blockIdx.x * blockDim.x + threadIdx.x;
  if (idx >= nE * 4) return;
  int e = idx >> 2, h = idx & 3;
  int dN = dst[e];
  float m = dec_f(amax[dN * 4 + h]);
  float ea = expf(a_ea[idx] - m);
  a_ea[idx] = ea;
  atomicAdd(&esum[dN * 4 + h], ea);
}

// ---------- message + aggregate: agg[dst] += alpha * (xw[src] + te @ Wt), WMMA ----------
__global__ __launch_bounds__(256) void message_kernel(
    const int* __restrict__ src, const int* __restrict__ dst, const float* __restrict__ ts,
    const float* __restrict__ freq, const float* __restrict__ phase,
    const _Float16* __restrict__ WtSw, const float* __restrict__ xw,
    const float* __restrict__ ea, const float* __restrict__ esum,
    float* __restrict__ agg, int nE)
{
  __shared__ __align__(16) _Float16 sB[WSW_HALVES];  // 40KB
  __shared__ __align__(16) _Float16 sA[64 * LDA];    // 17.4KB
  __shared__ float sF[128], sP[128], sAl[64 * 4], sTs[64];
  __shared__ int sS[64], sD[64];
  int tid = threadIdx.x;
  int e0 = blockIdx.x * 64;
  const uint4* gB = (const uint4*)WtSw;
  uint4* lB = (uint4*)sB;
  for (int i = tid; i < WSW_U4; i += 256) lB[i] = gB[i];
  if (tid < 128) { sF[tid] = freq[tid]; sP[tid] = phase[tid]; }
  if (tid < 64) {
    int e = e0 + tid; bool v = e < nE;
    sS[tid] = v ? src[e] : 0;
    sD[tid] = v ? dst[e] : 0;
    sTs[tid] = v ? ts[e] : 0.f;
  }
  __syncthreads();
  {
    int le = tid >> 2, h = tid & 3;
    int e = e0 + le;
    float alpha = 0.f;
    if (e < nE) alpha = ea[e * 4 + h] / (esum[sD[le] * 4 + h] + 1e-16f);
    sAl[tid] = alpha;
  }
  for (int idx = tid; idx < 64 * 128; idx += 256) {
    int le = idx >> 7, t = idx & 127;
    sA[le * LDA + t] = (_Float16)cosf(sTs[le] * sF[t] + sP[t]);
  }
  __syncthreads();
  int w = tid >> 5, lane = tid & 31;
  int rt = w & 3, ct0 = (w >> 2) * 4;
  int n = lane & 15, hi = (lane >> 4) & 1;
  const bool full = (e0 + 64 <= nE);   // block-uniform tail guard (E%64==0 -> always true)
  const _Float16* aBase = sA + (rt * 16 + n) * LDA + hi * 8;
  v16h afr[4];
#pragma unroll
  for (int tk = 0; tk < 4; ++tk) afr[tk] = load_a_frag(aBase + tk * 32);
  for (int cti = 0; cti < 4; ++cti) {
    int ct = ct0 + cti;
    v8f acc = {};
#pragma unroll
    for (int tk = 0; tk < 4; ++tk) {
      v16h b = load_b_frag(sB, tk * 8 + ct, lane);
      acc = __builtin_amdgcn_wmma_f32_16x16x32_f16(false, afr[tk], false, b, (short)0, acc, false, false);
    }
    int col = ct * 16 + n;
    int h = col >> 5;
    if (full) {
#pragma unroll
      for (int q = 0; q < 8; ++q) {
        int le = rt * 16 + q + hi * 8;
        float alpha = sAl[le * 4 + h];
        float hj = xw[(size_t)sS[le] * 128 + col] + acc[q];
        atomicAdd(&agg[(size_t)sD[le] * 128 + col], alpha * hj);
      }
    } else {
#pragma unroll
      for (int q = 0; q < 8; ++q) {
        int le = rt * 16 + q + hi * 8;
        if (e0 + le < nE) {
          float alpha = sAl[le * 4 + h];
          float hj = xw[(size_t)sS[le] * 128 + col] + acc[q];
          atomicAdd(&agg[(size_t)sD[le] * 128 + col], alpha * hj);
        }
      }
    }
  }
}

// ---------- finalize: out = agg + bias (optional leaky between layers) ----------
__global__ void finalize_kernel(const float* __restrict__ agg, const float* __restrict__ bias,
                                float* __restrict__ out, int count, int leaky) {
  int idx = blockIdx.x * blockDim.x + threadIdx.x;
  if (idx >= count) return;
  float v = agg[idx] + bias[idx & 127];
  if (leaky) v = (v > 0.f) ? v : v * SLOPE;
  out[idx] = v;
}

extern "C" void kernel_launch(void* const* d_in, const int* in_sizes, int n_in,
                              void* d_out, int out_size, void* d_ws, size_t ws_size,
                              hipStream_t stream) {
  const float* x     = (const float*)d_in[0];
  const int*   eidx  = (const int*)d_in[1];
  const float* ts    = (const float*)d_in[2];
  const float* freq  = (const float*)d_in[3];
  const float* phase = (const float*)d_in[4];
  const float* W0    = (const float*)d_in[5];
  const float* attl0 = (const float*)d_in[6];
  const float* attr0 = (const float*)d_in[7];
  const float* bias0 = (const float*)d_in[8];
  const float* W1    = (const float*)d_in[9];
  const float* attl1 = (const float*)d_in[10];
  const float* attr1 = (const float*)d_in[11];
  const float* bias1 = (const float*)d_in[12];

  const int N = in_sizes[0] / 128;   // 20000
  const int E = in_sizes[2];         // 640000
  const int* src = eidx;
  const int* dst = eidx + E;

  char* p = (char*)d_ws;
  auto carve = [&](size_t bytes) -> void* {
    void* r = (void*)p;
    p += (bytes + 255) & ~(size_t)255;
    return r;
  };
  _Float16* WxSw0 = (_Float16*)carve(WSW_HALVES * sizeof(_Float16));
  _Float16* WtSw0 = (_Float16*)carve(WSW_HALVES * sizeof(_Float16));
  _Float16* WxSw1 = (_Float16*)carve(WSW_HALVES * sizeof(_Float16));
  _Float16* WtSw1 = (_Float16*)carve(WSW_HALVES * sizeof(_Float16));
  float* vl0  = (float*)carve(512 * sizeof(float));
  float* vl1  = (float*)carve(512 * sizeof(float));
  float* c00  = (float*)carve(4 * sizeof(float));
  float* c01  = (float*)carve(4 * sizeof(float));
  float* xw   = (float*)carve((size_t)N * 128 * sizeof(float));
  float* al   = (float*)carve((size_t)N * 4 * sizeof(float));
  float* ar   = (float*)carve((size_t)N * 4 * sizeof(float));
  float* a_ea = (float*)carve((size_t)E * 4 * sizeof(float));
  unsigned* amax = (unsigned*)carve((size_t)N * 4 * sizeof(unsigned));
  float* esum = (float*)carve((size_t)N * 4 * sizeof(float));
  float* agg  = (float*)carve((size_t)N * 128 * sizeof(float));
  float* hfeat = (float*)carve((size_t)N * 128 * sizeof(float));

  const int eblocks = (E + 63) / 64;
  const int nblocks = (N + 63) / 64;
  const int nthr4   = (N * 4 + 255) / 256;
  const int ethr4   = (E * 4 + 255) / 256;
  const int nthr128 = (N * 128 + 255) / 256;

  prep_kernel<<<1, 128, 0, stream>>>(W0, attl0, attr0, phase, WxSw0, WtSw0, vl0, c00);
  prep_kernel<<<1, 128, 0, stream>>>(W1, attl1, attr1, phase, WxSw1, WtSw1, vl1, c01);

  // ---- layer 0 ----
  zero3_kernel<<<nthr128, 256, 0, stream>>>(amax, esum, agg, N);
  gemm_feat_kernel<<<nblocks, 256, 0, stream>>>(x, WxSw0, xw, N);
  attn_node_kernel<<<nthr4, 256, 0, stream>>>(xw, attl0, attr0, al, ar, N);
  edge_logits_kernel<<<eblocks, 256, 0, stream>>>(src, dst, ts, freq, phase, vl0, c00, al, ar, a_ea, amax, E);
  edge_exp_kernel<<<ethr4, 256, 0, stream>>>(a_ea, dst, amax, esum, E);
  message_kernel<<<eblocks, 256, 0, stream>>>(src, dst, ts, freq, phase, WtSw0, xw, a_ea, esum, agg, E);
  finalize_kernel<<<nthr128, 256, 0, stream>>>(agg, bias0, hfeat, N * 128, 1);

  // ---- layer 1 ----
  zero3_kernel<<<nthr128, 256, 0, stream>>>(amax, esum, agg, N);
  gemm_feat_kernel<<<nblocks, 256, 0, stream>>>(hfeat, WxSw1, xw, N);
  attn_node_kernel<<<nthr4, 256, 0, stream>>>(xw, attl1, attr1, al, ar, N);
  edge_logits_kernel<<<eblocks, 256, 0, stream>>>(src, dst, ts, freq, phase, vl1, c01, al, ar, a_ea, amax, E);
  edge_exp_kernel<<<ethr4, 256, 0, stream>>>(a_ea, dst, amax, esum, E);
  message_kernel<<<eblocks, 256, 0, stream>>>(src, dst, ts, freq, phase, WtSw1, xw, a_ea, esum, agg, E);
  finalize_kernel<<<nthr128, 256, 0, stream>>>(agg, bias1, (float*)d_out, N * 128, 0);
}
